// LSTM_3624952398014
// MI455X (gfx1250) — matrix operand order repaired and checked by reading, not compiled, not run on hardware
//
#include <hip/hip_runtime.h>

typedef __attribute__((ext_vector_type(16))) _Float16 v16h;
typedef __attribute__((ext_vector_type(8)))  _Float16 v8h;
typedef __attribute__((ext_vector_type(8)))  float    v8f;

#define B_    64
#define T_    2048
#define F_    128
#define H_    512
#define G_    2048
#define KTOT  640          // H + F fused K dimension
#define NWG   32           // one WG per 16-wide hidden slice
#define WGSZ  128          // 4 waves of 32 (wave32)
#define LROWS 64           // 4 gates x 16 gate-columns staged per WG
#define LSTRIDE 648        // padded LDS row stride (f16) to avoid bank conflicts

// ---------------- prep: pack W_cat = [W_hh | W_ih] to f16, fuse biases, zero state ----
__global__ void lstm_prep(const float* __restrict__ Whh, const float* __restrict__ Wih,
                          const float* __restrict__ bih, const float* __restrict__ bhh,
                          _Float16* __restrict__ Wcat, float* __restrict__ bsum,
                          _Float16* __restrict__ hbuf, unsigned* __restrict__ bar) {
    int idx = blockIdx.x * blockDim.x + threadIdx.x;
    if (idx < G_ * KTOT) {
        int g = idx / KTOT, k = idx % KTOT;
        float v = (k < H_) ? Whh[g * H_ + k] : Wih[g * F_ + (k - H_)];
        Wcat[idx] = (_Float16)v;
    }
    if (idx < G_) bsum[idx] = bih[idx] + bhh[idx];
    if (idx < 2 * B_ * H_) hbuf[idx] = (_Float16)0.f;   // h0 = 0 (both ping-pong bufs)
    if (idx == 0) *bar = 0u;
}

__device__ __forceinline__ float sigf(float x)   { return 1.f / (1.f + __expf(-x)); }
// tanh via single v_exp_f32: exact limits at +/-inf, ample accuracy for gate range
__device__ __forceinline__ float tanh_fast(float x) {
    return 1.f - 2.f / (__expf(2.f * x) + 1.f);
}

// ---------------- persistent fused LSTM layer-0 scan + output head -------------------
__global__ __launch_bounds__(WGSZ, 1) void lstm_persist(
    const float* __restrict__ x, const _Float16* __restrict__ Wcat,
    const float* __restrict__ bsum, _Float16* __restrict__ hbuf,
    const float* __restrict__ Wout, const float* __restrict__ bout,
    float* __restrict__ out, unsigned* __restrict__ bar) {
    extern __shared__ _Float16 sW[];           // 64 rows x 648 f16 = 82,944 B
    const int w   = blockIdx.x;                // hidden slice [16w, 16w+16)
    const int tid = threadIdx.x;
    const int wv  = tid >> 5;                  // batch row-tile 0..3 (16 rows each)
    const int L   = tid & 31;                  // lane in wave32
    const int Lr  = L & 15;                    // row/col within 16x16 tile
    const int Lh  = L >> 4;                    // lane-half select

    // Stage this WG's 64 W_cat rows (4 gates x 16 cols, full K=640) into LDS once,
    // using the CDNA5 async DMA path: global -> LDS with no VGPR staging.
    {
        const unsigned long long wbase = (unsigned long long)Wcat;
        for (int i = tid; i < LROWS * (KTOT / 8); i += WGSZ) {
            int lr = i / (KTOT / 8);
            int kc = (i % (KTOT / 8)) * 8;
            int gr = (lr >> 4) * H_ + 16 * w + (lr & 15);       // global gate row
            unsigned goff = (unsigned)(gr * KTOT + kc) * 2u;    // byte offset, 16B aligned
            unsigned ldsa = (unsigned)(uintptr_t)&sW[lr * LSTRIDE + kc];
            asm volatile("global_load_async_to_lds_b128 %0, %1, %2"
                         :: "v"(ldsa), "v"(goff), "s"(wbase) : "memory");
        }
        asm volatile("s_wait_asynccnt 0" ::: "memory");
    }
    __syncthreads();

    const int hcol = 16 * w + Lr;              // this lane's hidden column (C/D N=lane)
    float bs[4];
    #pragma unroll
    for (int g = 0; g < 4; ++g) bs[g] = bsum[g * H_ + hcol];

    v8f c = {};                                // cell-state tile: lives in VGPRs for all T
    const int brow = wv * 16 + Lr;             // A-matrix row (batch index) for this lane
    const size_t xrow0 = (size_t)brow * ((size_t)T_ * F_);

    for (int t = 0; t < T_; ++t) {
        const _Float16* hcur = hbuf + (size_t)(t & 1) * (B_ * H_);
        _Float16*       hnxt = hbuf + (size_t)((t + 1) & 1) * (B_ * H_);
        const float*    xrow = x + xrow0 + (size_t)t * F_;
        __builtin_prefetch(xrow + F_, 0, 1);   // global_prefetch x_{t+1}

        v8f acc[4];
        #pragma unroll
        for (int g = 0; g < 4; ++g)
            #pragma unroll
            for (int e = 0; e < 8; ++e) acc[g][e] = bs[g];

        // ---- recurrent part: K = 0..511 from h_t (f16, global/L2) -------------------
        #pragma unroll 4
        for (int kb = 0; kb < 16; ++kb) {
            const int k0 = kb * 32;
            const int ka = k0 + 8 * Lh;
            v8h alo = *(const v8h*)&hcur[brow * H_ + ka];
            v8h ahi = *(const v8h*)&hcur[brow * H_ + ka + 16];
            v16h a;
            #pragma unroll
            for (int e = 0; e < 8; ++e) { a[e] = alo[e]; a[e + 8] = ahi[e]; }
            const int kbb = k0 + 8 * Lh;
            #pragma unroll
            for (int g = 0; g < 4; ++g) {
                const _Float16* bp = &sW[(g * 16 + Lr) * LSTRIDE + kbb];
                v8h blo = *(const v8h*)bp;
                v8h bhi = *(const v8h*)(bp + 16);
                v16h bm;
                #pragma unroll
                for (int e = 0; e < 8; ++e) { bm[e] = blo[e]; bm[e + 8] = bhi[e]; }
                acc[g] = __builtin_amdgcn_wmma_f32_16x16x32_f16(
                    false, a, false, bm, (short)0, acc[g], false, false);
            }
        }
        // ---- input part: K = 512..639 from x_t (f32 -> f16 in-register) -------------
        #pragma unroll
        for (int kb = 0; kb < 4; ++kb) {
            const int k0 = kb * 32;
            const int ka = k0 + 8 * Lh;
            v16h a;
            #pragma unroll
            for (int e = 0; e < 8; ++e) {
                a[e]     = (_Float16)xrow[ka + e];
                a[e + 8] = (_Float16)xrow[ka + 16 + e];
            }
            const int kbb = H_ + k0 + 8 * Lh;
            #pragma unroll
            for (int g = 0; g < 4; ++g) {
                const _Float16* bp = &sW[(g * 16 + Lr) * LSTRIDE + kbb];
                v8h blo = *(const v8h*)bp;
                v8h bhi = *(const v8h*)(bp + 16);
                v16h bm;
                #pragma unroll
                for (int e = 0; e < 8; ++e) { bm[e] = blo[e]; bm[e + 8] = bhi[e]; }
                acc[g] = __builtin_amdgcn_wmma_f32_16x16x32_f16(
                    false, a, false, bm, (short)0, acc[g], false, false);
            }
        }

        // ---- elementwise LSTM cell (fp32, c resident in VGPRs) ----------------------
        #pragma unroll
        for (int e = 0; e < 8; ++e) {
            float ig = sigf(acc[0][e]);
            float fg = sigf(acc[1][e]);
            float gg = tanh_fast(acc[2][e]);
            float og = sigf(acc[3][e]);
            float cv = fg * c[e] + ig * gg;
            c[e] = cv;
            float hv = og * tanh_fast(cv);
            int row = wv * 16 + e + 8 * Lh;    // C/D layout: lanes 16-31 hold M=e+8
            hnxt[row * H_ + hcol] = (_Float16)hv;
        }

        // ---- grid-wide step barrier (monotonic count, re-zeroed by prep kernel) -----
        __threadfence();
        __syncthreads();
        if (tid == 0) {
            __hip_atomic_fetch_add(bar, 1u, __ATOMIC_ACQ_REL, __HIP_MEMORY_SCOPE_AGENT);
            const unsigned tgt = (unsigned)NWG * (unsigned)(t + 1);
            while (__hip_atomic_load(bar, __ATOMIC_ACQUIRE, __HIP_MEMORY_SCOPE_AGENT) < tgt)
                __builtin_amdgcn_s_sleep(1);
        }
        __syncthreads();
        __threadfence();
    }

    // ---- output head: out = relu(h_T) @ W_out^T + b_out (h_T is in hbuf[0]) --------
    if (w == 0) {
        const _Float16* hT = hbuf;             // T even -> final h in buffer 0
        for (int idx = tid; idx < B_ * 4; idx += WGSZ) {
            int b = idx >> 2, o = idx & 3;
            float s = bout[o];
            for (int j = 0; j < H_; ++j) {
                float hv = (float)hT[b * H_ + j];
                hv = hv > 0.f ? hv : 0.f;
                s += hv * Wout[o * H_ + j];
            }
            out[idx] = s;
        }
    }
}

extern "C" void kernel_launch(void* const* d_in, const int* in_sizes, int n_in,
                              void* d_out, int out_size, void* d_ws, size_t ws_size,
                              hipStream_t stream) {
    (void)in_sizes; (void)n_in; (void)out_size; (void)ws_size;
    const float* x    = (const float*)d_in[0];
    const float* Wih  = (const float*)d_in[1];
    const float* Whh  = (const float*)d_in[2];
    const float* bih  = (const float*)d_in[3];
    const float* bhh  = (const float*)d_in[4];
    const float* Wout = (const float*)d_in[9];
    const float* bout = (const float*)d_in[10];
    float* out = (float*)d_out;

    char* ws = (char*)d_ws;
    _Float16* Wcat = (_Float16*)ws;                              // 2048*640*2  = 2,621,440 B
    float*    bsum = (float*)(ws + 2621440);                     // 2048*4      =     8,192 B
    _Float16* hbuf = (_Float16*)(ws + 2621440 + 8192);           // 2*64*512*2  =   131,072 B
    unsigned* bar  = (unsigned*)(ws + 2621440 + 8192 + 131072);  // barrier counter

    const int n = G_ * KTOT;
    lstm_prep<<<(n + 255) / 256, 256, 0, stream>>>(Whh, Wih, bih, bhh, Wcat, bsum, hbuf, bar);

    const size_t lds = (size_t)LROWS * LSTRIDE * sizeof(_Float16);  // 82,944 B < 320 KB
    lstm_persist<<<NWG, WGSZ, lds, stream>>>(x, Wcat, bsum, hbuf, Wout, bout, out, bar);
}
